// LongMemKVCache_69088843923762
// MI455X (gfx1250) — compile-verified, matrix operand order checked by reading, not run on hardware
//
#include <hip/hip_runtime.h>
#include <hip/hip_bf16.h>

typedef __attribute__((ext_vector_type(16))) __bf16 v16bf;
typedef __attribute__((ext_vector_type(8)))  __bf16 v8bf;
typedef __attribute__((ext_vector_type(8)))  float  v8f;
typedef __attribute__((ext_vector_type(4)))  unsigned int v4u;
typedef __attribute__((ext_vector_type(8)))  int    v8i;
typedef __attribute__((ext_vector_type(4)))  int    v4i;

#define B_Q   1024
#define N_K   131072
#define DK    256
#define DV    256
#define TOPK  8
#define PARTS 32
#define KEYS_PER_PART (N_K / PARTS)   // 4096 keys per partition
#define TILE  64                       // keys per block iteration (4 waves x 16)
#define NEG_INF (-__builtin_inff())

// Dynamic-LDS layout (no static __shared__ in the hot kernel, so the dynamic
// segment starts at LDS offset 0 and the TDM descriptor lds_addr is exact):
//   [0, 16384)      : key tile buffer 0   (64 rows x 128 bf16)
//   [16384, 32768)  : key tile buffer 1
//   [32768, 36864)  : score tile (4 waves x 16 x 16 f32)
#define KBUF_BYTES 16384
#define LDS_BYTES  36864

// ---------------------------------------------------------------------------
// Kernel 1a: L2-normalize queries -> bf16 (folds 1/||q|| into the A operand)
// ---------------------------------------------------------------------------
__global__ void qnorm_kernel(const float* __restrict__ q, __bf16* __restrict__ qn) {
    __shared__ float red[256];
    const int b = blockIdx.x, t = threadIdx.x;
    const float v = q[(size_t)b * DK + t];
    red[t] = v * v;
    __syncthreads();
    for (int s = 128; s > 0; s >>= 1) {
        if (t < s) red[t] += red[t + s];
        __syncthreads();
    }
    const float inv = 1.0f / fmaxf(sqrtf(red[0]), 1e-12f);
    qn[(size_t)b * DK + t] = (__bf16)(v * inv);
}

// ---------------------------------------------------------------------------
// Kernel 1b: L2-normalize keys -> bf16 copy (kbf). One HBM pass; afterwards
// the 64 MB bf16 key matrix is L2-resident and the hot loop needs no
// conversions and no per-column scaling.
// ---------------------------------------------------------------------------
__global__ void knorm_convert_kernel(const float* __restrict__ k,
                                     __bf16* __restrict__ kbf) {
    __shared__ float red[256];
    const int n = blockIdx.x, t = threadIdx.x;
    const float v = k[(size_t)n * DK + t];
    red[t] = v * v;
    __syncthreads();
    for (int s = 128; s > 0; s >>= 1) {
        if (t < s) red[t] += red[t + s];
        __syncthreads();
    }
    const float inv = 1.0f / fmaxf(sqrtf(red[0]), 1e-12f);
    kbf[(size_t)n * DK + t] = (__bf16)(v * inv);
}

// ---------------------------------------------------------------------------
// Kernel 2: fused bf16-WMMA similarity + per-partition top-8, with TDM
// double-buffered staging of key tiles into LDS.
// Block = 128 threads (4 waves); each wave owns a 16-key column tile.
// Each 64-key tile is staged in two half-K (128-element) TDM transfers so the
// double buffer fits in 36 KB of LDS; wave 0 issues stage s+1 while all waves
// compute on stage s (4 WMMAs per stage, 8 per 64-key tile).
// A (16x32 bf16) ISA layout: lanes 0-15 -> M=lane, K={kk+0..7, kk+16..23};
// lanes 16-31 -> K={kk+8..15, kk+24..31}. B mirrors the K striping with the
// lane indexing column N. D (v8f): lane%16 = N, VGPR r -> M = r + 8*(lane>=16).
// ---------------------------------------------------------------------------
__global__ __launch_bounds__(128) void simtopk_kernel(
    const __bf16* __restrict__ kbf, const __bf16* __restrict__ qn,
    const unsigned char* __restrict__ valid,
    float* __restrict__ pval, int* __restrict__ pidx) {

    extern __shared__ char smem_raw[];
    __bf16* ktile  = (__bf16*)smem_raw;                 // 2 x (64 x 128) bf16
    float*  sscore = (float*)(smem_raw + 2 * KBUF_BYTES);

    const int qtile = blockIdx.x;
    const int part  = blockIdx.y;
    const int tid   = threadIdx.x;
    const int wave  = tid >> 5;
    const int lane  = tid & 31;
    const int lhalf = lane >> 4;            // 0: low 16 lanes, 1: high 16 lanes
    const int l16   = lane & 15;
    const int c0    = lhalf ? 8 : 0;        // K-striping offset per lane half

    const __bf16* qbase = qn + (size_t)(qtile * 16 + l16) * DK;
    const int part_base = part * KEYS_PER_PART;

    // TDM issue: stage a (64 rows x 128 bf16) half-K key tile into LDS buffer.
    const unsigned long long kb_base = (unsigned long long)(size_t)kbf;
    auto tdm_issue = [&](int row0, int half, int buf) {
        const unsigned long long gaddr =
            kb_base + ((unsigned long long)(part_base + row0) * DK +
                       (unsigned long long)half * 128ull) * 2ull;
        v4u g0;
        g0[0] = 1u;                                     // count=1, no gather
        g0[1] = (unsigned)(buf * KBUF_BYTES);           // lds_addr (bytes)
        g0[2] = (unsigned)(gaddr & 0xffffffffu);        // global_addr[31:0]
        g0[3] = (unsigned)(gaddr >> 32) | (2u << 30);   // global_addr[56:32] | type=2
        v8i g1;
        g1[0] = 1 << 16;              // workgroup_mask=0, data_size=1 (2 bytes)
        g1[1] = (int)(128u << 16);    // tensor_dim0 = 128 (bits 79:48 lo)
        g1[2] = (int)(64u << 16);     // tensor_dim0 hi=0, tensor_dim1 = 64
        g1[3] = (int)(128u << 16);    // tensor_dim1 hi=0, tile_dim0 = 128
        g1[4] = 64;                   // tile_dim1 = 64, tile_dim2 = 0
        g1[5] = 256;                  // tensor_dim0_stride = 256 elems (row pitch)
        g1[6] = 0;                    // stride hi, tensor_dim1_stride lo
        g1[7] = 0;
        v4i g2 = {0, 0, 0, 0};        // 2D tensor: groups 2/3 unused
        v4i g3 = {0, 0, 0, 0};
        v8i g4 = {0, 0, 0, 0, 0, 0, 0, 0};
        __builtin_amdgcn_tensor_load_to_lds(g0, g1, g2, g3, g4, 0);
    };

    // Running sorted top-8 (descending) held by merge threads 0..15 (row = tid).
    float tv[TOPK];
    int   ti[TOPK];
    if (tid < 16) {
        for (int j = 0; j < TOPK; ++j) { tv[j] = NEG_INF; ti[j] = 0; }
    }

    const int NSTAGE = (KEYS_PER_PART / TILE) * 2;      // 128 half-K stages
    if (wave == 0) tdm_issue(0, 0, 0);                  // prologue: stage 0

    int stage = 0;
    for (int it = 0; it < KEYS_PER_PART; it += TILE) {
        v8f acc = {};
        for (int half = 0; half < 2; ++half, ++stage) {
            if (wave == 0) __builtin_amdgcn_s_wait_tensorcnt(0);
            __syncthreads();                            // stage data visible
            if (wave == 0) {
                const int ns = stage + 1;
                if (ns < NSTAGE) tdm_issue((ns >> 1) * TILE, ns & 1, ns & 1);
            }
            const int buf = stage & 1;
            const __bf16* kb = ktile + buf * (KBUF_BYTES / 2)
                                     + (wave * 16 + l16) * 128;
#pragma unroll
            for (int kk = 0; kk < 128; kk += 32) {
                const int gk = half * 128 + kk + c0;
                // ---- A fragment (loop-invariant over keys; hoisted) ----
                v8bf alo = *(const v8bf*)(qbase + gk);
                v8bf ahi = *(const v8bf*)(qbase + gk + 16);
                v16bf a;
#pragma unroll
                for (int i = 0; i < 8; ++i) { a[i] = alo[i]; a[8 + i] = ahi[i]; }
                // ---- B fragment from the staged LDS tile ----
                v8bf blo = *(const v8bf*)(kb + kk + c0);
                v8bf bhi = *(const v8bf*)(kb + kk + c0 + 16);
                v16bf bf;
#pragma unroll
                for (int i = 0; i < 8; ++i) { bf[i] = blo[i]; bf[8 + i] = bhi[i]; }

                acc = __builtin_amdgcn_wmma_f32_16x16x32_bf16(
                    false, a, false, bf, (short)0, acc, false, false);
            }
            __syncthreads();    // buf fully consumed before stage+2 refills it
        }

        // Validity mask (keys pre-normalized, no scaling); spill tile to LDS.
        const int  krow = part_base + it + wave * 16 + l16;
        const bool vld  = valid[krow] != 0;
#pragma unroll
        for (int r = 0; r < 8; ++r) {
            const float s = vld ? acc[r] : NEG_INF;
            sscore[wave * 256 + (r + lhalf * 8) * 16 + l16] = s;
        }
        __syncthreads();

        // 16 merge threads: row tid consumes 64 candidates with threshold gate.
        if (tid < 16) {
            for (int w = 0; w < 4; ++w) {
                const int base_n = part_base + it + w * 16;
                for (int n = 0; n < 16; ++n) {
                    const float s = sscore[w * 256 + tid * 16 + n];
                    if (s > tv[TOPK - 1]) {
                        int pos = TOPK - 1;
                        while (pos > 0 && tv[pos - 1] < s) {
                            tv[pos] = tv[pos - 1];
                            ti[pos] = ti[pos - 1];
                            --pos;
                        }
                        tv[pos] = s;
                        ti[pos] = base_n + n;
                    }
                }
            }
        }
        __syncthreads();
    }

    if (tid < 16) {
        const int    b    = qtile * 16 + tid;
        const size_t base = ((size_t)b * PARTS + part) * TOPK;
        for (int j = 0; j < TOPK; ++j) {
            pval[base + j] = tv[j];
            pidx[base + j] = ti[j];
        }
    }
}

// ---------------------------------------------------------------------------
// Kernel 3: merge the 32 partitions' top-8 per query, gather key/value rows.
// ---------------------------------------------------------------------------
__global__ void merge_gather_kernel(const float* __restrict__ keys,
                                    const float* __restrict__ values,
                                    const float* __restrict__ pval,
                                    const int* __restrict__ pidx,
                                    float* __restrict__ outK,
                                    float* __restrict__ outV) {
    __shared__ int lidx[TOPK];
    const int b = blockIdx.x, t = threadIdx.x;
    if (t == 0) {
        float tv[TOPK]; int ti[TOPK];
        for (int j = 0; j < TOPK; ++j) { tv[j] = NEG_INF; ti[j] = 0; }
        const size_t base = (size_t)b * PARTS * TOPK;
        for (int c = 0; c < PARTS * TOPK; ++c) {
            const float s = pval[base + c];
            if (s > tv[TOPK - 1]) {
                int pos = TOPK - 1;
                while (pos > 0 && tv[pos - 1] < s) {
                    tv[pos] = tv[pos - 1];
                    ti[pos] = ti[pos - 1];
                    --pos;
                }
                tv[pos] = s;
                ti[pos] = pidx[base + c];
            }
        }
        for (int j = 0; j < TOPK; ++j) lidx[j] = ti[j];
    }
    __syncthreads();
    for (int j = 0; j < TOPK; ++j) {
        const int id = lidx[j];
        outK[((size_t)b * TOPK + j) * DK + t] = keys[(size_t)id * DK + t];
        outV[((size_t)b * TOPK + j) * DV + t] = values[(size_t)id * DV + t];
    }
}

// ---------------------------------------------------------------------------
// Launcher. Inputs: 0=query[B,256] f32, 1=keys[N,256] f32, 2=values[N,256] f32,
// 3=valid[N] bool, 4=top_k (scalar, fixed 8). Output: keys gather then values
// gather, each [1024,8,256] f32, concatenated flat.
// Workspace: qn (512 KB) | kbf (64 MB bf16 normalized keys) | pval | pidx.
// ---------------------------------------------------------------------------
extern "C" void kernel_launch(void* const* d_in, const int* in_sizes, int n_in,
                              void* d_out, int out_size, void* d_ws, size_t ws_size,
                              hipStream_t stream) {
    const float*         query  = (const float*)d_in[0];
    const float*         keys   = (const float*)d_in[1];
    const float*         values = (const float*)d_in[2];
    const unsigned char* valid  = (const unsigned char*)d_in[3];

    char* ws = (char*)d_ws;
    size_t off = 0;
    __bf16* qn  = (__bf16*)(ws + off); off += (size_t)B_Q * DK * 2;     // 512 KB
    __bf16* kbf = (__bf16*)(ws + off); off += (size_t)N_K * DK * 2;     // 64 MB
    float*  pval = (float*)(ws + off); off += (size_t)B_Q * PARTS * TOPK * 4;
    int*    pidx = (int*)(ws + off);

    float* outK = (float*)d_out;
    float* outV = (float*)d_out + (size_t)B_Q * TOPK * DK;

    qnorm_kernel<<<B_Q, 256, 0, stream>>>(query, qn);
    knorm_convert_kernel<<<N_K, 256, 0, stream>>>(keys, kbf);
    simtopk_kernel<<<dim3(B_Q / 16, PARTS), 128, LDS_BYTES, stream>>>(
        kbf, qn, valid, pval, pidx);
    merge_gather_kernel<<<B_Q, 256, 0, stream>>>(keys, values, pval, pidx, outK, outV);
}
